// DMRLayer_26508538150957
// MI455X (gfx1250) — matrix operand order repaired and checked by reading, not compiled, not run on hardware
//
#include <hip/hip_runtime.h>

// ============================================================================
// DMR layer for MI455X (gfx1250, wave32).
// Roofline: ~140 MB of gathers/intermediates -> ~6us @ 23.3 TB/s HBM.
// ~5 GFLOP of matmul would be VALU-bound in fp32; routed through
// v_wmma_f32_16x16x32_f16 (f32 accum) so the kernel is gather/HBM-bound.
// ============================================================================

typedef __attribute__((ext_vector_type(16))) _Float16 v16h;
typedef __attribute__((ext_vector_type(8)))  _Float16 v8h;
typedef __attribute__((ext_vector_type(8)))  float    v8f;

#define DEV __device__ __forceinline__

constexpr int BATCH = 8192;
constexpr int L     = 50;
constexpr int NNEG  = 20;
constexpr int E     = 16;
constexpr int XF    = 256;   // MLP input 226 padded to 256 (8 K-chunks of 32)
constexpr int H1P   = 224;   // hidden1 200 padded to 224 (7 K-chunks of 32)

// -------- workspace layout (float offsets into d_ws) ------------------------
constexpr int OFF_POS_I2I = 0;                      // [50][48] f32
constexpr int OFF_POS_U2I = 2400;                   // [50][48] f32
constexpr int OFF_S       = 4800;                   // 1 f32 (padded)
constexpr int OFF_PLOSS   = 4864;                   // [B] f32
constexpr int OFF_NEGS    = OFF_PLOSS + BATCH;      // [B] f32
constexpr int OFF_WET     = OFF_NEGS + BATCH;       // 2x[64][64] f16 (= 4096 f32)
constexpr int OFF_XFH     = OFF_WET + 4096;         // [B][256] f16
constexpr int OFF_H1H     = OFF_XFH + BATCH*XF/2;   // [B][224] f16
// total ~ 8 MB

// -------- WMMA helpers ------------------------------------------------------
DEV v8f wmma16(v16h a, v16h b, v8f c) {
  // 8 args: (neg_a, A, neg_b, B, c_mod, C, reuse_a, reuse_b)
  return __builtin_amdgcn_wmma_f32_16x16x32_f16(false, a, false, b,
                                                (short)0, c, false, false);
}

// A fragment 16x32 f16: base = tile row0 + k-chunk col offset already applied.
// lane<16: m=lane, VGPR v<4 -> k=2v,2v+1 ; v>=4 -> k=16+...; lanes 16-31 shift k by 8.
DEV v16h load_a(const _Float16* base, int stride, int lane) {
  const _Float16* p = base + (lane & 15) * stride + ((lane >> 4) << 3);
  v8h lo = *(const v8h*)p;
  v8h hi = *(const v8h*)(p + 16);
  v16h a;
#pragma unroll
  for (int i = 0; i < 8; ++i) { a[i] = lo[i]; a[i + 8] = hi[i]; }
  return a;
}

// B fragment 32x16 f16 from n-major [N][K] storage: lane n = lane&15,
// same k order as load_a: v<4 -> k=2v,2v+1 ; v>=4 -> k=16+...; lanes 16-31 shift k by 8.
DEV v16h load_b(const _Float16* baseN, int strideK, int lane) {
  const _Float16* p = baseN + (lane & 15) * strideK + ((lane >> 4) << 3);
  v8h lo = *(const v8h*)p;
  v8h hi = *(const v8h*)(p + 16);
  v16h b;
#pragma unroll
  for (int i = 0; i < 8; ++i) { b[i] = lo[i]; b[i + 8] = hi[i]; }
  return b;
}

DEV float wredsum(float v) {
#pragma unroll
  for (int o = 16; o > 0; o >>= 1) v += __shfl_xor(v, o, 32);
  return v;
}
DEV float wredmax(float v) {
#pragma unroll
  for (int o = 16; o > 0; o >>= 1) v = fmaxf(v, __shfl_xor(v, o, 32));
  return v;
}

// X_series[64x48 padded] @ We[48x48] via 4x3x2 = 24 WMMAs, result -> hid f32
DEV void series_matmul(const _Float16* XsH, const _Float16* WeT,
                       float* hid, int lane) {
#pragma unroll
  for (int mt = 0; mt < 4; ++mt) {
#pragma unroll
    for (int nt = 0; nt < 3; ++nt) {
      v8f acc = {};
#pragma unroll
      for (int kc = 0; kc < 2; ++kc) {
        v16h a = load_a(XsH + mt * 16 * 64 + kc * 32, 64, lane);
        v16h b = load_b(WeT + nt * 16 * 64 + kc * 32, 64, lane);
        acc = wmma16(a, b, acc);
      }
      // C/D layout: n = lane&15, m = r + 8*(lane>>4)
      int n  = nt * 16 + (lane & 15);
      int mb = mt * 16 + ((lane >> 4) << 3);
#pragma unroll
      for (int r = 0; r < 8; ++r) hid[(mb + r) * 48 + n] = acc[r];
    }
  }
}

// ============================================================================
// K0: batch-invariant precompute: pos@Wp (f32) and WeT (f16, padded, n-major)
// ============================================================================
struct PreParams {
  const float *pos, *Wpi, *Wpu, *Wei, *Weu;
  float* ws;
};
__global__ __launch_bounds__(256) void k_pre(PreParams p) {
  int gt = blockIdx.x * 256 + threadIdx.x;
  for (int i = gt; i < 2 * 2400; i += 8192) {
    int which = i / 2400, r = i % 2400, l = r / 48, h = r % 48;
    const float* Wp = which ? p.Wpu : p.Wpi;
    float acc = 0.f;
    for (int e = 0; e < 48; ++e) acc += p.pos[l * 48 + e] * Wp[e * 48 + h];
    p.ws[(which ? OFF_POS_U2I : OFF_POS_I2I) + l * 48 + h] = acc;
  }
  _Float16* wet = (_Float16*)(p.ws + OFF_WET);
  for (int i = gt; i < 2 * 64 * 64; i += 8192) {
    int which = i / 4096, r = i % 4096, n = r >> 6, k = r & 63;
    const float* We = which ? p.Weu : p.Wei;
    float v = (n < 48 && k < 48) ? We[k * 48 + n] : 0.f;  // WeT[n][k] = We[k][n]
    wet[i] = (_Float16)v;
  }
}

// ============================================================================
// K1: fused gather + dual attention (WMMA) + aux losses. 1 wave per row.
// ============================================================================
struct AttParams {
  const int *cate, *ser, *neg, *len;
  const float *emb, *Wc, *zi, *zu;
  float* ws;
};
__global__ __launch_bounds__(32) void k_attention(AttParams p) {
  __shared__ __align__(32) _Float16 XsH[64 * 64];  // series f16, zero-padded
  __shared__ float hid[64 * 48];
  __shared__ float Xng[NNEG * 48];
  __shared__ float Xit[48], iWc[48];
  __shared__ float sS[64], aA[64];
  __shared__ float uo[48], ov[48], lvv[48];

  const int b = blockIdx.x, lane = threadIdx.x;
  float* ws = p.ws;
  const _Float16* WeTi = (const _Float16*)(ws + OFF_WET);
  const _Float16* WeTu = WeTi + 64 * 64;
  _Float16* xfh = (_Float16*)(ws + OFF_XFH) + (size_t)b * XF;

  for (int i = lane; i < 64 * 64; i += 32) XsH[i] = (_Float16)0.f;
  __syncthreads();

  // gather series embeddings (3 ids x 16 dims per step -> 48-wide rows)
  for (int idx = lane; idx < L * 3; idx += 32) {
    int id = p.ser[b * L * 3 + idx];
    int l = idx / 3, pp = idx % 3;
    const float* e = p.emb + (size_t)id * E;
#pragma unroll
    for (int k = 0; k < E; ++k) XsH[l * 64 + pp * 16 + k] = (_Float16)e[k];
  }
  for (int idx = lane; idx < NNEG * 3; idx += 32) {
    int id = p.neg[b * NNEG * 3 + idx];
    int l = idx / 3, pp = idx % 3;
    const float* e = p.emb + (size_t)id * E;
#pragma unroll
    for (int k = 0; k < E; ++k) Xng[l * 48 + pp * 16 + k] = e[k];
  }
  // cate features (all 8) -> xfeat[0..127]; last 3 also form X_item
  for (int idx = lane; idx < 128; idx += 32) {
    int j = idx >> 4, k = idx & 15;
    int id = p.cate[b * 8 + j];
    float v = p.emb[(size_t)id * E + k];
    xfh[idx] = (_Float16)v;
    if (j >= 5) Xit[(j - 5) * 16 + k] = v;
  }
  for (int i = lane; i < XF - 226; i += 32) xfh[226 + i] = (_Float16)0.f;
  __syncthreads();

  // itemWc[h] = X_item @ i2i_Wc
  for (int h = lane; h < 48; h += 32) {
    float acc = 0.f;
    for (int e = 0; e < 48; ++e) acc += Xit[e] * p.Wc[e * 48 + h];
    iWc[h] = acc;
  }
  __syncthreads();

  const int len = p.len[b];

  // -------------------- I2I attention --------------------
  series_matmul(XsH, WeTi, hid, lane);
  __syncthreads();
  {
    const float* posW = ws + OFF_POS_I2I;
    for (int l = lane; l < L; l += 32) {
      float s = 0.f;
      for (int h = 0; h < 48; ++h)
        s += p.zi[h] * tanhf(hid[l * 48 + h] + posW[l * 48 + h] + iWc[h]);
      sS[l] = s;
    }
  }
  __syncthreads();
  float mx = -1e30f;
  for (int l = lane; l < len; l += 32) mx = fmaxf(mx, sS[l]);
  mx = wredmax(mx);
  float es = 0.f, rawsum = 0.f;
  for (int l = lane; l < len; l += 32) {
    float e = expf(sS[l] - mx);
    aA[l] = e; es += e; rawsum += sS[l];
  }
  es = wredsum(es); rawsum = wredsum(rawsum);
  float inv = 1.f / es;
  __syncthreads();
  for (int l = lane; l < 64; l += 32) { float v = (l < len) ? aA[l] * inv : 0.f; aA[l] = v; }
  __syncthreads();
  for (int e = lane; e < 48; e += 32) {
    float acc = 0.f;
    for (int l = 0; l < len; ++l) acc += aA[l] * (float)XsH[l * 64 + e];
    xfh[128 + e] = (_Float16)acc;
  }
  if (lane == 0) xfh[176] = (_Float16)rawsum;
  __syncthreads();

  // -------------------- U2I attention --------------------
  series_matmul(XsH, WeTu, hid, lane);
  __syncthreads();
  {
    const float* posW = ws + OFF_POS_U2I;
    for (int l = lane; l < L; l += 32) {
      float s = 0.f;
      for (int h = 0; h < 48; ++h)
        s += p.zu[h] * tanhf(hid[l * 48 + h] + posW[l * 48 + h]);
      sS[l] = s;
    }
  }
  __syncthreads();
  mx = -1e30f;
  for (int l = lane; l < len; l += 32) mx = fmaxf(mx, sS[l]);
  mx = wredmax(mx);
  es = 0.f;
  for (int l = lane; l < len; l += 32) { float e = expf(sS[l] - mx); aA[l] = e; es += e; }
  es = wredsum(es);
  inv = 1.f / es;
  __syncthreads();
  for (int l = lane; l < 64; l += 32) { float v = (l < len) ? aA[l] * inv : 0.f; aA[l] = v; }
  __syncthreads();
  for (int e = lane; e < 48; e += 32) {
    float acc = 0.f;
    for (int l = 0; l < len; ++l) acc += aA[l] * (float)XsH[l * 64 + e];
    float lastv = aA[len - 1] * (float)XsH[(len - 1) * 64 + e];
    uo[e] = acc; lvv[e] = lastv; ov[e] = acc - lastv;  // out_vec = sum_{l<len-1}
    xfh[177 + e] = (_Float16)acc;
  }
  __syncthreads();

  // inner product + aux cosine losses
  float dIn = 0.f, dLO = 0.f, nL = 0.f, nO = 0.f;
  for (int e = lane; e < 48; e += 32) {
    dIn += uo[e] * Xit[e];
    dLO += lvv[e] * ov[e];
    nL  += lvv[e] * lvv[e];
    nO  += ov[e] * ov[e];
  }
  dIn = wredsum(dIn); dLO = wredsum(dLO); nL = wredsum(nL); nO = wredsum(nO);
  if (lane == 0) xfh[225] = (_Float16)dIn;

  float invL = rsqrtf(fmaxf(nL, 1e-12f));
  float invO = rsqrtf(fmaxf(nO, 1e-12f));
  float cpos = dLO * invL * invO;
  float ploss = log1pf(expf(-(1.f - cpos) * 0.5f));   // softplus(-pos_logit)

  float nl = 0.f;
  if (lane < NNEG) {
    float d = 0.f, nn = 0.f;
    for (int e = 0; e < 48; ++e) { float x = Xng[lane * 48 + e]; d += x * ov[e]; nn += x * x; }
    float c = d * rsqrtf(fmaxf(nn, 1e-12f)) * invO;
    nl = log1pf(expf((1.f - c) * 0.5f));              // softplus(neg_logit)
  }
  nl = wredsum(nl);
  if (lane == 0) { ws[OFF_PLOSS + b] = ploss; ws[OFF_NEGS + b] = nl; }
}

// ============================================================================
// K2: global reduction of pos_loss over the batch (TF sums pos over its axis)
// ============================================================================
__global__ __launch_bounds__(256) void k_reduce(float* ws) {
  __shared__ float sm[256];
  float s = 0.f;
  for (int i = threadIdx.x; i < BATCH; i += 256) s += ws[OFF_PLOSS + i];
  sm[threadIdx.x] = s;
  __syncthreads();
  for (int o = 128; o > 0; o >>= 1) {
    if (threadIdx.x < o) sm[threadIdx.x] += sm[threadIdx.x + o];
    __syncthreads();
  }
  if (threadIdx.x == 0) ws[OFF_S] = sm[0];
}

// ============================================================================
// K3: MLP layer 1: x[16rows,256K] @ W1[226,200] (+b1, LN eps=1e-3, PReLU)
// 256 threads = 8 waves; wave w owns N-tiles {w, 8+w} of 13.
// ============================================================================
struct Mlp1Params {
  const float *W1, *b1, *g1, *be1, *a1;
  float* ws;
};
__global__ __launch_bounds__(256) void k_mlp1(Mlp1Params p) {
  __shared__ __align__(32) _Float16 Xt[16 * 256];   // 8 KB
  __shared__ __align__(32) _Float16 Wt[208 * 32];   // 13 KB, n-major chunk
  __shared__ float hd[16 * 208];                    // 13 KB
  __shared__ float mstat[16], rstat[16];

  const int t = threadIdx.x, lane = t & 31, w = t >> 5;
  const int rowbase = blockIdx.x * 16;
  const _Float16* xfh = (const _Float16*)(p.ws + OFF_XFH) + (size_t)rowbase * XF;
  _Float16* h1h = (_Float16*)(p.ws + OFF_H1H) + (size_t)rowbase * H1P;

  for (int i = t; i < 16 * 256; i += 256) Xt[i] = xfh[i];

  v8f acc0 = {}, acc1 = {};
  const int nt0 = w, nt1 = 8 + w;
  for (int kc = 0; kc < 8; ++kc) {
    for (int i = t; i < 208 * 32; i += 256) {
      int n = i >> 5, kk = i & 31, k = kc * 32 + kk;
      Wt[i] = (k < 226 && n < 200) ? (_Float16)p.W1[k * 200 + n] : (_Float16)0.f;
    }
    if (t == 0 && kc < 7) __builtin_prefetch(p.W1 + (kc + 1) * 32 * 200, 0, 1);
    __syncthreads();
    v16h a  = load_a(Xt + kc * 32, 256, lane);
    v16h b0 = load_b(Wt + nt0 * 16 * 32, 32, lane);
    acc0 = wmma16(a, b0, acc0);
    if (nt1 < 13) {                       // wave-uniform; EXEC stays all-ones
      v16h b1f = load_b(Wt + nt1 * 16 * 32, 32, lane);
      acc1 = wmma16(a, b1f, acc1);
    }
    __syncthreads();
  }

  // store tiles (+bias) to LDS
  {
    int mb = (lane >> 4) << 3;
    int n = nt0 * 16 + (lane & 15);
    float bias = (n < 200) ? p.b1[n] : 0.f;
#pragma unroll
    for (int r = 0; r < 8; ++r) hd[(mb + r) * 208 + n] = acc0[r] + bias;
    if (nt1 < 13) {
      int n1 = nt1 * 16 + (lane & 15);
      float bias1 = (n1 < 200) ? p.b1[n1] : 0.f;
#pragma unroll
      for (int r = 0; r < 8; ++r) hd[(mb + r) * 208 + n1] = acc1[r] + bias1;
    }
  }
  __syncthreads();
  if (t < 16) {  // LayerNorm stats over 200
    float s = 0.f, sq = 0.f;
    for (int n = 0; n < 200; ++n) { float v = hd[t * 208 + n]; s += v; sq += v * v; }
    float m = s / 200.f;
    mstat[t] = m;
    rstat[t] = rsqrtf(sq / 200.f - m * m + 1e-3f);
  }
  __syncthreads();
  for (int i = t; i < 16 * H1P; i += 256) {
    int m = i / H1P, n = i % H1P;
    float v = 0.f;
    if (n < 200) {
      float x = p.g1[n] * (hd[m * 208 + n] - mstat[m]) * rstat[m] + p.be1[n];
      v = x > 0.f ? x : p.a1[n] * x;      // PReLU
    }
    h1h[i] = (_Float16)v;
  }
}

// ============================================================================
// K4: MLP layer 2 + head: h1[16,224K] @ W2[200,80] (+b2,LN,PReLU),
//     pred = sigmoid(h2 @ W3 + b3); out = (pred, S + negsum)
// ============================================================================
struct Mlp2Params {
  const float *W2, *b2, *g2, *be2, *a2, *W3, *b3;
  float* ws;
  float* out;
};
__global__ __launch_bounds__(256) void k_mlp2(Mlp2Params p) {
  __shared__ __align__(32) _Float16 Ht[16 * 224];   // 7 KB
  __shared__ __align__(32) _Float16 Wt[80 * 32];    // 5 KB
  __shared__ float hd[16 * 80];
  __shared__ float mstat[16], rstat[16];
  __shared__ float w3s[80];

  const int t = threadIdx.x, lane = t & 31, w = t >> 5;
  const int rowbase = blockIdx.x * 16;
  const _Float16* h1h = (const _Float16*)(p.ws + OFF_H1H) + (size_t)rowbase * H1P;

  for (int i = t; i < 16 * 224; i += 256) Ht[i] = h1h[i];
  for (int i = t; i < 80; i += 256) w3s[i] = p.W3[i];

  v8f acc = {};
  for (int kc = 0; kc < 7; ++kc) {
    for (int i = t; i < 80 * 32; i += 256) {
      int n = i >> 5, kk = i & 31, k = kc * 32 + kk;
      Wt[i] = (k < 200) ? (_Float16)p.W2[k * 80 + n] : (_Float16)0.f;
    }
    __syncthreads();
    if (w < 5) {                          // waves 0..4 own the 5 N-tiles
      v16h a = load_a(Ht + kc * 32, 224, lane);
      v16h b = load_b(Wt + w * 16 * 32, 32, lane);
      acc = wmma16(a, b, acc);
    }
    __syncthreads();
  }
  if (w < 5) {
    int mb = (lane >> 4) << 3;
    int n = w * 16 + (lane & 15);
    float bias = p.b2[n];
#pragma unroll
    for (int r = 0; r < 8; ++r) hd[(mb + r) * 80 + n] = acc[r] + bias;
  }
  __syncthreads();
  if (t < 16) {
    float s = 0.f, sq = 0.f;
    for (int n = 0; n < 80; ++n) { float v = hd[t * 80 + n]; s += v; sq += v * v; }
    float m = s / 80.f;
    mstat[t] = m;
    rstat[t] = rsqrtf(sq / 80.f - m * m + 1e-3f);
  }
  __syncthreads();
  for (int i = t; i < 16 * 80; i += 256) {
    int m = i / 80, n = i % 80;
    float x = p.g2[n] * (hd[i] - mstat[m]) * rstat[m] + p.be2[n];
    hd[i] = x > 0.f ? x : p.a2[n] * x;
  }
  __syncthreads();
  if (t < 16) {
    float d = p.b3[0];
    for (int n = 0; n < 80; ++n) d += hd[t * 80 + n] * w3s[n];
    float pred = 1.f / (1.f + expf(-d));
    int row = rowbase + t;
    p.out[row * 2 + 0] = pred;
    p.out[row * 2 + 1] = p.ws[OFF_S] + p.ws[OFF_NEGS + row];
  }
}

// ============================================================================
extern "C" void kernel_launch(void* const* d_in, const int* in_sizes, int n_in,
                              void* d_out, int out_size, void* d_ws, size_t ws_size,
                              hipStream_t stream) {
  (void)in_sizes; (void)n_in; (void)out_size; (void)ws_size;

  const int*   cate = (const int*)d_in[0];
  const int*   ser  = (const int*)d_in[1];
  const int*   neg  = (const int*)d_in[2];
  const int*   len  = (const int*)d_in[3];
  const float* emb  = (const float*)d_in[4];
  const float* pos  = (const float*)d_in[5];
  const float* Wc   = (const float*)d_in[6];
  const float* Wpi  = (const float*)d_in[7];
  const float* Wei  = (const float*)d_in[8];
  const float* zi   = (const float*)d_in[9];
  const float* Wpu  = (const float*)d_in[10];
  const float* Weu  = (const float*)d_in[11];
  const float* zu   = (const float*)d_in[12];
  const float* W1   = (const float*)d_in[13];
  const float* b1   = (const float*)d_in[14];
  const float* g1   = (const float*)d_in[15];
  const float* be1  = (const float*)d_in[16];
  const float* a1   = (const float*)d_in[17];
  const float* W2   = (const float*)d_in[18];
  const float* b2   = (const float*)d_in[19];
  const float* g2   = (const float*)d_in[20];
  const float* be2  = (const float*)d_in[21];
  const float* a2   = (const float*)d_in[22];
  const float* W3   = (const float*)d_in[23];
  const float* b3   = (const float*)d_in[24];

  float* ws  = (float*)d_ws;
  float* out = (float*)d_out;

  PreParams pp{pos, Wpi, Wpu, Wei, Weu, ws};
  k_pre<<<dim3(32), dim3(256), 0, stream>>>(pp);

  AttParams ap{cate, ser, neg, len, emb, Wc, zi, zu, ws};
  k_attention<<<dim3(BATCH), dim3(32), 0, stream>>>(ap);

  k_reduce<<<dim3(1), dim3(256), 0, stream>>>(ws);

  Mlp1Params m1{W1, b1, g1, be1, a1, ws};
  k_mlp1<<<dim3(BATCH / 16), dim3(256), 0, stream>>>(m1);

  Mlp2Params m2{W2, b2, g2, be2, a2, W3, b3, ws, out};
  k_mlp2<<<dim3(BATCH / 16), dim3(256), 0, stream>>>(m2);
}
